// Attention_52458730553798
// MI455X (gfx1250) — matrix operand order repaired and checked by reading, not compiled, not run on hardware
//
#include <hip/hip_runtime.h>

// Bahdanau attention, fused for MI455X (gfx1250).
// B=32, S=2048, H=1024.
//   pre[b,h]  = q[b]@Wa + Wa_b + Ua_b                 (small, f32)
//   scores    = sum_h tanh(pre + key@Ua) * va          (bf16 WMMA GEMM, fused epilogue)
//   weights   = softmax_S(scores)                      (va_b cancels -> dropped)
//   context   = weights @ key                          (memory bound, f32 atomics)

#define B_ 32
#define S_ 2048
#define H_ 1024
#define LDA 1032   // 1024 bf16 + 8 pad -> row stride 2064B = 516 dwords, bank-conflict free b128 frags

typedef unsigned short u16;
typedef __attribute__((ext_vector_type(16))) __bf16 v16bf;
typedef __attribute__((ext_vector_type(8)))  float  v8f;
typedef __attribute__((ext_vector_type(4)))  float  v4f;
typedef __attribute__((ext_vector_type(4)))  unsigned int v4u;
typedef __attribute__((ext_vector_type(2)))  unsigned int v2u;

union Frag { v4u q[2]; v16bf v; };

__device__ __forceinline__ u16 f2bf(float f) {           // round-to-nearest-even f32->bf16
  unsigned int u = __float_as_uint(f);
  u += 0x7FFFu + ((u >> 16) & 1u);
  return (u16)(u >> 16);
}

__device__ __forceinline__ float tanh_hw(float x) {
  // CDNA5 hardware transcendental; v_nop covers the TRANS-op result hazard.
  float y;
  asm volatile("v_tanh_f32 %0, %1\n\tv_nop" : "=v"(y) : "v"(x));
  return y;
}

__device__ __forceinline__ v8f wmma_bf16(v16bf a, v16bf b, v8f c) {
  // (neg_a, A, neg_b, B, c_mod, C, reuse_a, reuse_b)
  return __builtin_amdgcn_wmma_f32_16x16x32_bf16(false, a, false, b, (short)0, c, false, false);
}

// ---------------- kernel 1a: Ua (H,H) f32 row-major -> UaT (n-major, K contiguous) bf16 ----
__global__ __launch_bounds__(256) void convert_ua_kernel(const float* __restrict__ Ua,
                                                         u16* __restrict__ UaT) {
  int idx = blockIdx.x * 256 + threadIdx.x;   // coalesced read of Ua
  int k = idx >> 10, n = idx & (H_ - 1);
  UaT[(size_t)n * H_ + k] = f2bf(Ua[idx]);
}

// ---------------- kernel 1b: pre[b,h] = q[b]@Wa + Wa_b + Ua_b -----------------------------
__global__ __launch_bounds__(256) void pre_kernel(const float* __restrict__ q,
                                                  const float* __restrict__ Wa,
                                                  const float* __restrict__ Wab,
                                                  const float* __restrict__ Uab,
                                                  float* __restrict__ pre) {
  __shared__ float qS[H_];
  const int b = blockIdx.x, tid = threadIdx.x;
  #pragma unroll
  for (int j = 0; j < 4; ++j) qS[tid + j * 256] = q[b * H_ + tid + j * 256];
  __syncthreads();
  float acc[4];
  #pragma unroll
  for (int j = 0; j < 4; ++j) { int h = tid + j * 256; acc[j] = Wab[h] + Uab[h]; }
  for (int k = 0; k < H_; ++k) {
    const float qv = qS[k];
    const float* wrow = Wa + (size_t)k * H_;
    #pragma unroll
    for (int j = 0; j < 4; ++j) acc[j] += qv * wrow[tid + j * 256];
  }
  #pragma unroll
  for (int j = 0; j < 4; ++j) pre[b * H_ + tid + j * 256] = acc[j];
}

// ---------------- kernel 2: fused WMMA energies + score reduction -------------------------
// grid (S/64, B), 256 threads (8 waves). WG: 64 S-rows x all 1024 N-cols, K=1024.
__global__ __launch_bounds__(256) void energies_kernel(
    const float* __restrict__ key, const u16* __restrict__ UaT,
    const float* __restrict__ pre, const float* __restrict__ va,
    float* __restrict__ scores)
{
  extern __shared__ char smem[];
  u16*   As   = (u16*)smem;                    // 64 x LDA bf16 (132 KB)
  float* preS = (float*)(smem + 64 * LDA * 2); // 1024
  float* vaS  = preS + H_;                     // 1024
  float* scS  = vaS + H_;                      // 64

  const int b   = blockIdx.y;
  const int s0  = blockIdx.x * 64;
  const int tid = threadIdx.x;

  for (int i = tid; i < H_; i += 256) { preS[i] = pre[b * H_ + i]; vaS[i] = va[i]; }
  if (tid < 64) scS[tid] = 0.0f;

  // Stage 64 key rows, f32 -> bf16, into padded LDS (one float4 per thread per row).
  for (int row = 0; row < 64; ++row) {
    const v4f* src = (const v4f*)(key + ((size_t)(s0 + row) * B_ + b) * H_);
    v4f x = src[tid];
    v2u pk;
    pk.x = (unsigned)f2bf(x.x) | ((unsigned)f2bf(x.y) << 16);
    pk.y = (unsigned)f2bf(x.z) | ((unsigned)f2bf(x.w) << 16);
    *(v2u*)(As + row * LDA + tid * 4) = pk;
  }
  __syncthreads();

  const int wv   = tid >> 5;
  const int lane = tid & 31;
  const int lhi  = lane >> 4;   // high half-wave
  const int l15  = lane & 15;

  float rp[4][8];               // per-lane row partial scores (M = m*16 + r + 8*lhi)
  #pragma unroll
  for (int m = 0; m < 4; ++m)
    #pragma unroll
    for (int r = 0; r < 8; ++r) rp[m][r] = 0.0f;

  for (int p = 0; p < 4; ++p) {
    const int pi = wv + p * 8;                                     // 32-col group 0..31
    const u16* bp0 = UaT + (size_t)(pi * 32 + l15) * H_ + (lhi ? 16 : 0);
    const u16* bp1 = bp0 + (size_t)16 * H_;

    v8f acc[4][2];
    #pragma unroll
    for (int m = 0; m < 4; ++m)
      #pragma unroll
      for (int j = 0; j < 2; ++j)
        #pragma unroll
        for (int r = 0; r < 8; ++r) acc[m][j][r] = 0.0f;

    for (int kk = 0; kk < H_; kk += 32) {
      Frag b0, b1;                                                 // B frags: K contiguous per lane
      b0.q[0] = *(const v4u*)(bp0 + kk);
      b0.q[1] = *(const v4u*)(bp0 + kk + 8);
      b1.q[0] = *(const v4u*)(bp1 + kk);
      b1.q[1] = *(const v4u*)(bp1 + kk + 8);
      #pragma unroll
      for (int m = 0; m < 4; ++m) {
        Frag a;                                                    // A frag: K contiguous per lane, same order as B
        const u16* ap = As + (m * 16 + l15) * LDA + kk + (lhi ? 16 : 0);
        a.q[0] = *(const v4u*)(ap);        // K = base .. base+7
        a.q[1] = *(const v4u*)(ap + 8);    // K = base+8 .. base+15
        acc[m][0] = wmma_bf16(a.v, b0.v, acc[m][0]);
        acc[m][1] = wmma_bf16(a.v, b1.v, acc[m][1]);
      }
    }

    // Fused epilogue: e = tanh(acc + pre[n]); score += e * va[n]
    #pragma unroll
    for (int m = 0; m < 4; ++m)
      #pragma unroll
      for (int j = 0; j < 2; ++j) {
        const int n = pi * 32 + j * 16 + l15;
        const float pv = preS[n], vv = vaS[n];
        #pragma unroll
        for (int r = 0; r < 8; ++r)
          rp[m][r] += tanh_hw(acc[m][j][r] + pv) * vv;
      }
  }

  #pragma unroll
  for (int m = 0; m < 4; ++m)
    #pragma unroll
    for (int r = 0; r < 8; ++r)
      atomicAdd(&scS[m * 16 + r + lhi * 8], rp[m][r]);             // ds_add_f32

  __syncthreads();
  if (tid < 64) scores[b * S_ + s0 + tid] = scS[tid];
}

// ---------------- kernel 3: softmax over S per batch; also zero context slice -------------
__global__ __launch_bounds__(256) void softmax_kernel(const float* __restrict__ scores,
                                                      float* __restrict__ out) {
  __shared__ float red[256];
  const int b = blockIdx.x, tid = threadIdx.x;
  float v[8];
  float mx = -3.4e38f;
  #pragma unroll
  for (int j = 0; j < 8; ++j) { v[j] = scores[b * S_ + tid + j * 256]; mx = fmaxf(mx, v[j]); }
  red[tid] = mx; __syncthreads();
  for (int st = 128; st > 0; st >>= 1) { if (tid < st) red[tid] = fmaxf(red[tid], red[tid + st]); __syncthreads(); }
  const float m = red[0]; __syncthreads();
  float sum = 0.0f;
  #pragma unroll
  for (int j = 0; j < 8; ++j) { v[j] = __expf(v[j] - m); sum += v[j]; }
  red[tid] = sum; __syncthreads();
  for (int st = 128; st > 0; st >>= 1) { if (tid < st) red[tid] += red[tid + st]; __syncthreads(); }
  const float inv = 1.0f / red[0];
  #pragma unroll
  for (int j = 0; j < 8; ++j) out[B_ * H_ + b * S_ + tid + j * 256] = v[j] * inv;
  #pragma unroll
  for (int j = 0; j < 4; ++j) out[b * H_ + tid + j * 256] = 0.0f;   // zero context for atomics
}

// ---------------- kernel 4: context[b,h] = sum_s weights[b,s] * key[s,b,h] ----------------
__global__ __launch_bounds__(256) void context_kernel(const float* __restrict__ key,
                                                      const float* __restrict__ weights,
                                                      float* __restrict__ ctx) {
  __shared__ float wS[64];
  const int b = blockIdx.y, s0 = blockIdx.x * 64, tid = threadIdx.x;
  if (tid < 64) wS[tid] = weights[b * S_ + s0 + tid];
  __syncthreads();
  float acc[4] = {0.f, 0.f, 0.f, 0.f};
  for (int s = 0; s < 64; ++s) {
    const float w = wS[s];
    const float* row = key + ((size_t)(s0 + s) * B_ + b) * H_;
    #pragma unroll
    for (int j = 0; j < 4; ++j) acc[j] += w * row[tid + j * 256];
  }
  #pragma unroll
  for (int j = 0; j < 4; ++j) atomicAdd(&ctx[b * H_ + tid + j * 256], acc[j]);
}

extern "C" void kernel_launch(void* const* d_in, const int* in_sizes, int n_in,
                              void* d_out, int out_size, void* d_ws, size_t ws_size,
                              hipStream_t stream) {
  const float* query = (const float*)d_in[0];
  const float* key   = (const float*)d_in[1];
  const float* Wa_w  = (const float*)d_in[2];
  const float* Wa_b  = (const float*)d_in[3];
  const float* Ua_w  = (const float*)d_in[4];
  const float* Ua_b  = (const float*)d_in[5];
  const float* va_w  = (const float*)d_in[6];
  // d_in[7] = va_b: cancels under softmax and context; mathematically exact to drop.
  (void)in_sizes; (void)n_in; (void)out_size;
  float* out = (float*)d_out;

  char* ws = (char*)d_ws;
  u16*   UaT    = (u16*)ws;                                        // 2 MB bf16 Ua^T
  float* pre    = (float*)(ws + (size_t)H_ * H_ * 2);              // 128 KB
  float* scores = (float*)(ws + (size_t)H_ * H_ * 2 + (size_t)B_ * H_ * 4); // 256 KB
  (void)ws_size;

  convert_ua_kernel<<<(H_ * H_) / 256, 256, 0, stream>>>(Ua_w, UaT);
  pre_kernel<<<B_, 256, 0, stream>>>(query, Wa_w, Wa_b, Ua_b, pre);

  const size_t smem = (size_t)64 * LDA * 2 + H_ * 4 + H_ * 4 + 64 * 4; // ~140.5 KB (CDNA5: 320 KB/WGP)
  energies_kernel<<<dim3(S_ / 64, B_), 256, smem, stream>>>(key, UaT, pre, va_w, scores);

  softmax_kernel<<<B_, 256, 0, stream>>>(scores, out);
  context_kernel<<<dim3(S_ / 64, B_), 256, 0, stream>>>(key, out + B_ * H_, out);
}
